// Pointnet2Backbone_27436251087200
// MI455X (gfx1250) — matrix operand order repaired and checked by reading, not compiled, not run on hardware
//
#include <hip/hip_runtime.h>
#include <hip/hip_bf16.h>

typedef __attribute__((ext_vector_type(16))) _Float16 v16h;
typedef __attribute__((ext_vector_type(8)))  _Float16 v8h;
typedef __attribute__((ext_vector_type(4)))  _Float16 v4h;
typedef __attribute__((ext_vector_type(8)))  float    v8f;

#define NPTS    20000
#define NCTR    2048
#define BATCH   2
#define NSAMPLE 64
#define NTOT    (BATCH * NPTS)          // 40000
#define RADIUS  0.04f
#define INV_R   25.0f

// ---------------------------------------------------------------------------
// 1) Farthest point sampling: one persistent 1024-thread WG per batch.
//    Points + running min-dist live in registers (20 slots/thread).
// ---------------------------------------------------------------------------
__global__ __launch_bounds__(1024) void fps_kernel(const float* __restrict__ xyz,
                                                   int* __restrict__ inds,
                                                   float* __restrict__ new_xyz) {
  const int b = blockIdx.x;
  const int t = threadIdx.x;
  const float* p = xyz + (size_t)b * NPTS * 3;
  float px[20], py[20], pz[20], md[20];
#pragma unroll
  for (int k = 0; k < 20; ++k) {
    int i = t + (k << 10);
    if (i < NPTS) {
      px[k] = p[3 * i + 0]; py[k] = p[3 * i + 1]; pz[k] = p[3 * i + 2];
      md[k] = 1e10f;
    } else {
      px[k] = 0.f; py[k] = 0.f; pz[k] = 0.f; md[k] = -1.f; // invalid slot
    }
  }
  __shared__ float s_best[1024];
  __shared__ int   s_bidx[1024];
  __shared__ float s_cur[3];

  int cur = 0;
  for (int s = 0; s < NCTR; ++s) {
    if (t == (cur & 1023)) {
      int k = cur >> 10;
      s_cur[0] = px[k]; s_cur[1] = py[k]; s_cur[2] = pz[k];
    }
    __syncthreads();
    float cx = s_cur[0], cy = s_cur[1], cz = s_cur[2];
    if (t == 0) {
      inds[b * NCTR + s] = cur;
      size_t o = ((size_t)b * NCTR + s) * 3;
      new_xyz[o + 0] = cx; new_xyz[o + 1] = cy; new_xyz[o + 2] = cz;
    }
    float best = -2.f; int bidx = 0x7fffffff;
#pragma unroll
    for (int k = 0; k < 20; ++k) {
      float dx = px[k] - cx, dy = py[k] - cy, dz = pz[k] - cz;
      float d2 = dx * dx + dy * dy + dz * dz;
      if (md[k] >= 0.f) {
        md[k] = fminf(md[k], d2);
        int i = t + (k << 10);
        if (md[k] > best || (md[k] == best && i < bidx)) { best = md[k]; bidx = i; }
      }
    }
    s_best[t] = best; s_bidx[t] = bidx;
    __syncthreads();
    for (int off = 512; off > 0; off >>= 1) {
      if (t < off) {
        float ob = s_best[t + off]; int oi = s_bidx[t + off];
        if (ob > s_best[t] || (ob == s_best[t] && oi < s_bidx[t])) {
          s_best[t] = ob; s_bidx[t] = oi;
        }
      }
      __syncthreads();
    }
    cur = s_bidx[0];
    __syncthreads();
  }
}

// ---------------------------------------------------------------------------
// 2) Ball query: one wave per center; ordered chunk scan + ballot/prefix keeps
//    the lowest-index-first semantics of the reference.
// ---------------------------------------------------------------------------
__global__ __launch_bounds__(32) void ballquery_kernel(const float* __restrict__ xyz,
                                                       const float* __restrict__ new_xyz,
                                                       int* __restrict__ gidx) {
  const int b = blockIdx.y, c = blockIdx.x;
  const int lane = threadIdx.x;
  __shared__ int list[NSAMPLE];
  const float* p = xyz + (size_t)b * NPTS * 3;
  size_t co = ((size_t)b * NCTR + c) * 3;
  float cx = new_xyz[co], cy = new_xyz[co + 1], cz = new_xyz[co + 2];
  const float r2 = RADIUS * RADIUS;
  int cnt = 0;
  for (int base = 0; base < NPTS; base += 32) {
    int i = base + lane;
    float dx = p[3 * i + 0] - cx, dy = p[3 * i + 1] - cy, dz = p[3 * i + 2] - cz;
    bool hit = (dx * dx + dy * dy + dz * dz) < r2;
    unsigned long long m = __ballot(hit);
    int pos = cnt + __popcll(m & ((1ull << lane) - 1ull));
    if (hit && pos < NSAMPLE) list[pos] = i;
    cnt += __popcll(m);
    if (cnt >= NSAMPLE) break;
  }
  __syncthreads();
  int nvalid = cnt < NSAMPLE ? cnt : NSAMPLE;
  int first = (cnt > 0) ? list[0] : 0;
  for (int j = lane; j < NSAMPLE; j += 32) {
    int v = (j < nvalid) ? list[j] : first;
    gidx[((size_t)(b * NCTR + c)) * NSAMPLE + j] = v;
  }
}

// ---------------------------------------------------------------------------
// 3) SA1: fused group + MLP(3->64->64->128, ReLU) + max over 64 samples.
//    One wave per center, 2 samples per lane, weights in LDS.
// ---------------------------------------------------------------------------
__global__ __launch_bounds__(256) void sa1_kernel(const float* __restrict__ xyz,
                                                  const float* __restrict__ new_xyz,
                                                  const int* __restrict__ gidx,
                                                  const float* __restrict__ W1, const float* __restrict__ B1,
                                                  const float* __restrict__ W2, const float* __restrict__ B2,
                                                  const float* __restrict__ W3, const float* __restrict__ B3,
                                                  float* __restrict__ f1) {
  __shared__ float lw1[64 * 3], lb1[64], lw2[64 * 64], lb2[64], lw3[128 * 64], lb3[128];
  const int tid = threadIdx.x;
  for (int i = tid; i < 64 * 3;   i += 256) lw1[i] = W1[i];
  for (int i = tid; i < 64;       i += 256) lb1[i] = B1[i];
  for (int i = tid; i < 64 * 64;  i += 256) lw2[i] = W2[i];
  for (int i = tid; i < 64;       i += 256) lb2[i] = B2[i];
  for (int i = tid; i < 128 * 64; i += 256) lw3[i] = W3[i];
  for (int i = tid; i < 128;      i += 256) lb3[i] = B3[i];
  __syncthreads();

  const int b = blockIdx.y;
  const int c = blockIdx.x * 8 + (tid >> 5);
  const int lane = tid & 31;
  size_t co = ((size_t)b * NCTR + c) * 3;
  float cx = new_xyz[co], cy = new_xyz[co + 1], cz = new_xyz[co + 2];
  const float* p = xyz + (size_t)b * NPTS * 3;

  float h2[2][64];
#pragma unroll
  for (int sg = 0; sg < 2; ++sg) {
    int smp = lane + sg * 32;
    int pi = gidx[((size_t)(b * NCTR + c)) * NSAMPLE + smp];
    float gx = (p[3 * pi + 0] - cx) * INV_R;
    float gy = (p[3 * pi + 1] - cy) * INV_R;
    float gz = (p[3 * pi + 2] - cz) * INV_R;
    float h1[64];
#pragma unroll
    for (int j = 0; j < 64; ++j)
      h1[j] = fmaxf(lw1[j * 3] * gx + lw1[j * 3 + 1] * gy + lw1[j * 3 + 2] * gz + lb1[j], 0.f);
    for (int j = 0; j < 64; ++j) {
      float acc = lb2[j];
#pragma unroll 8
      for (int k = 0; k < 64; ++k) acc += lw2[j * 64 + k] * h1[k];
      h2[sg][j] = fmaxf(acc, 0.f);
    }
  }
  for (int o = 0; o < 128; ++o) {
    float a0 = lb3[o], a1 = lb3[o];
#pragma unroll 8
    for (int k = 0; k < 64; ++k) {
      float w = lw3[o * 64 + k];
      a0 += w * h2[0][k]; a1 += w * h2[1][k];
    }
    float m = fmaxf(fmaxf(a0, 0.f), fmaxf(a1, 0.f));
    for (int off = 16; off > 0; off >>= 1) m = fmaxf(m, __shfl_xor(m, off, 32));
    if (lane == 0) f1[((size_t)b * 128 + o) * NCTR + c] = m;
  }
}

// ---------------------------------------------------------------------------
// 4) 3-NN of every point among the 2048 centers (centers cached in LDS),
//    with inverse-distance weights (reference semantics).
// ---------------------------------------------------------------------------
__global__ __launch_bounds__(256) void knn3_kernel(const float* __restrict__ xyz,
                                                   const float* __restrict__ new_xyz,
                                                   int* __restrict__ idx3,
                                                   float* __restrict__ w3) {
  __shared__ float sc[NCTR * 3];
  const int b = blockIdx.y;
  for (int i = threadIdx.x; i < NCTR * 3; i += 256) sc[i] = new_xyz[(size_t)b * NCTR * 3 + i];
  __syncthreads();
  int n = blockIdx.x * 256 + threadIdx.x;
  if (n >= NPTS) return;
  size_t po = ((size_t)b * NPTS + n) * 3;
  float x = xyz[po], y = xyz[po + 1], z = xyz[po + 2];
  float d0 = 1e30f, d1 = 1e30f, d2 = 1e30f;
  int i0 = 0, i1 = 0, i2 = 0;
  for (int s = 0; s < NCTR; ++s) {
    float dx = x - sc[3 * s], dy = y - sc[3 * s + 1], dz = z - sc[3 * s + 2];
    float d = dx * dx + dy * dy + dz * dz;
    if (d < d0)      { d2 = d1; i2 = i1; d1 = d0; i1 = i0; d0 = d; i0 = s; }
    else if (d < d1) { d2 = d1; i2 = i1; d1 = d;  i1 = s; }
    else if (d < d2) { d2 = d;  i2 = s; }
  }
  float w0 = 1.f / (fmaxf(d0, 0.f) + 1e-8f);
  float w1 = 1.f / (fmaxf(d1, 0.f) + 1e-8f);
  float w2 = 1.f / (fmaxf(d2, 0.f) + 1e-8f);
  float ws = w0 + w1 + w2;
  idx3[po] = i0; idx3[po + 1] = i1; idx3[po + 2] = i2;
  w3[po] = w0 / ws; w3[po + 1] = w1 / ws; w3[po + 2] = w2 / ws;
}

// ---------------------------------------------------------------------------
// 5) Interpolate f1 onto all points; emit f16 activation matrix X (128 x 40000)
// ---------------------------------------------------------------------------
__global__ __launch_bounds__(256) void interp_kernel(const float* __restrict__ f1,
                                                     const int* __restrict__ idx3,
                                                     const float* __restrict__ w3,
                                                     _Float16* __restrict__ X) {
  int c = blockIdx.y;
  int ng = blockIdx.x * 256 + threadIdx.x;
  if (ng >= NTOT) return;
  int b = ng / NPTS, n = ng % NPTS;
  size_t base = ((size_t)b * NPTS + n) * 3;
  const float* f = f1 + ((size_t)b * 128 + c) * NCTR;
  float v = f[idx3[base]] * w3[base] + f[idx3[base + 1]] * w3[base + 1]
          + f[idx3[base + 2]] * w3[base + 2];
  X[(size_t)c * NTOT + ng] = (_Float16)v;
}

// ---------------------------------------------------------------------------
// 6) Weight conversion f32 -> f16 (weights stay hot in L2 across the GEMMs)
// ---------------------------------------------------------------------------
__global__ __launch_bounds__(256) void wconv_kernel(const float* __restrict__ W1,
                                                    const float* __restrict__ W2,
                                                    _Float16* __restrict__ W1h,
                                                    _Float16* __restrict__ W2h) {
  int i = blockIdx.x * 256 + threadIdx.x;
  if (i < 512 * 128) W1h[i] = (_Float16)W1[i];
  if (i < 512 * 512) W2h[i] = (_Float16)W2[i];
}

// ---------------------------------------------------------------------------
// 7) WMMA GEMM: Out = relu(A(MxK,f16) * B(Kx40000,f16) + bias).
//    Block 256 = 8 waves (4 along M x 2 along N) -> 64x32 output tile.
//    B tile staged in LDS *column-major* (k contiguous per column, row padded
//    to 40 halves) so each B fragment is two ds_load_b128. A fragment is two
//    contiguous v8h global loads per the ISA's swizzled 16-bit A layout.
//    Double-buffered LDS: one __syncthreads per K-step, prefetch overlaps WMMA.
// ---------------------------------------------------------------------------
template <int K, bool OUT_HALF>
__global__ __launch_bounds__(256) void gemm_kernel(const _Float16* __restrict__ A,
                                                   const _Float16* __restrict__ Bm,
                                                   const float* __restrict__ bias,
                                                   void* __restrict__ Out) {
  constexpr int STEPS = K / 32;
  const int tid = threadIdx.x;
  const int wave = tid >> 5, lane = tid & 31;
  const int wm = wave >> 1, wn = wave & 1;
  const int tileM = blockIdx.y * 64 + wm * 16;
  const int tileN = blockIdx.x * 32;
  const int half = lane >> 4;
  const int mrow = tileM + (lane & 15);
  const int nl   = wn * 16 + (lane & 15);
  const int nCol = tileN + nl;

  // [buf][column n][k] ; 40-half row: keeps 16B alignment, staggers banks
  __shared__ _Float16 sB[2][32][40];

  // Staging: thread t covers column (t&31), k-rows (t>>5)*4 .. +3.
  // Global reads are 64B-coalesced per k-row across each wave.
  const int sn = tid & 31;
  const int sk = (tid >> 5) * 4;
  const _Float16* bsrc = Bm + tileN + sn + (size_t)sk * NTOT;
  const _Float16* arow = A + (size_t)mrow * K;

  _Float16 r0 = bsrc[0];
  _Float16 r1 = bsrc[NTOT];
  _Float16 r2 = bsrc[2 * (size_t)NTOT];
  _Float16 r3 = bsrc[3 * (size_t)NTOT];

  v8f acc = {0.f, 0.f, 0.f, 0.f, 0.f, 0.f, 0.f, 0.f};
#pragma unroll 4
  for (int s = 0; s < STEPS; ++s) {
    { // commit staged registers -> LDS buffer s&1 (one ds_store_b64)
      v4h pack; pack[0] = r0; pack[1] = r1; pack[2] = r2; pack[3] = r3;
      *(v4h*)(&sB[s & 1][sn][sk]) = pack;
    }
    __syncthreads();
    if (s + 1 < STEPS) { // prefetch next tile; overlaps fragment loads + WMMA
      const _Float16* nsrc = bsrc + (size_t)(s + 1) * 32 * NTOT;
      r0 = nsrc[0];
      r1 = nsrc[NTOT];
      r2 = nsrc[2 * (size_t)NTOT];
      r3 = nsrc[3 * (size_t)NTOT];
    }
    // A fragment: per-lane K chunks [8*half..+7] and [16+8*half..+7]
    const _Float16* ak = arow + s * 32 + 8 * half;
    v8h alo = *(const v8h*)(ak);
    v8h ahi = *(const v8h*)(ak + 16);
    v16h a = __builtin_shufflevector(alo, ahi, 0, 1, 2, 3, 4, 5, 6, 7,
                                     8, 9, 10, 11, 12, 13, 14, 15);
    // B fragment: K chunks [8*half..+7] and [16+8*half..+7] of column nl (two ds_load_b128)
    const _Float16* bk = &sB[s & 1][nl][8 * half];
    v8h blo = *(const v8h*)(bk);
    v8h bhi = *(const v8h*)(bk + 16);
    v16h bf = __builtin_shufflevector(blo, bhi, 0, 1, 2, 3, 4, 5, 6, 7,
                                      8, 9, 10, 11, 12, 13, 14, 15);
    acc = __builtin_amdgcn_wmma_f32_16x16x32_f16(false, a, false, bf,
                                                 (short)0, acc, false, false);
  }
#pragma unroll
  for (int r = 0; r < 8; ++r) {
    int m = tileM + r + 8 * half;
    float v = fmaxf(acc[r] + bias[m], 0.f);
    if (OUT_HALF) {
      ((_Float16*)Out)[(size_t)m * NTOT + nCol] = (_Float16)v;
    } else {
      int bb = nCol / NPTS, nn = nCol % NPTS;
      ((float*)Out)[((size_t)bb * 512 + m) * NPTS + nn] = v;
    }
  }
}

// ---------------------------------------------------------------------------
extern "C" void kernel_launch(void* const* d_in, const int* in_sizes, int n_in,
                              void* d_out, int out_size, void* d_ws, size_t ws_size,
                              hipStream_t stream) {
  (void)in_sizes; (void)n_in; (void)out_size; (void)ws_size;
  const float* pc  = (const float*)d_in[0];
  const float* sW1 = (const float*)d_in[1];
  const float* sb1 = (const float*)d_in[2];
  const float* sW2 = (const float*)d_in[3];
  const float* sb2 = (const float*)d_in[4];
  const float* sW3 = (const float*)d_in[5];
  const float* sb3 = (const float*)d_in[6];
  const float* fW1 = (const float*)d_in[37];
  const float* fb1 = (const float*)d_in[38];
  const float* fW2 = (const float*)d_in[39];
  const float* fb2 = (const float*)d_in[40];

  char* ws = (char*)d_ws;
  size_t off = 0;
  auto alloc = [&](size_t bytes) -> void* {
    void* p = ws + off;
    off = (off + bytes + 255) & ~(size_t)255;
    return p;
  };
  float*    new_xyz = (float*)alloc((size_t)BATCH * NCTR * 3 * sizeof(float));
  int*      finds   = (int*)alloc((size_t)BATCH * NCTR * sizeof(int));
  int*      gidx    = (int*)alloc((size_t)BATCH * NCTR * NSAMPLE * sizeof(int));
  float*    f1      = (float*)alloc((size_t)BATCH * 128 * NCTR * sizeof(float));
  int*      idx3    = (int*)alloc((size_t)BATCH * NPTS * 3 * sizeof(int));
  float*    w3      = (float*)alloc((size_t)BATCH * NPTS * 3 * sizeof(float));
  _Float16* Xh      = (_Float16*)alloc((size_t)128 * NTOT * sizeof(_Float16));
  _Float16* W1h     = (_Float16*)alloc((size_t)512 * 128 * sizeof(_Float16));
  _Float16* W2h     = (_Float16*)alloc((size_t)512 * 512 * sizeof(_Float16));
  _Float16* Y1h     = (_Float16*)alloc((size_t)512 * NTOT * sizeof(_Float16));

  fps_kernel<<<BATCH, 1024, 0, stream>>>(pc, finds, new_xyz);
  ballquery_kernel<<<dim3(NCTR, BATCH), 32, 0, stream>>>(pc, new_xyz, gidx);
  sa1_kernel<<<dim3(NCTR / 8, BATCH), 256, 0, stream>>>(pc, new_xyz, gidx,
                                                        sW1, sb1, sW2, sb2, sW3, sb3, f1);
  knn3_kernel<<<dim3((NPTS + 255) / 256, BATCH), 256, 0, stream>>>(pc, new_xyz, idx3, w3);
  wconv_kernel<<<(512 * 512 + 255) / 256, 256, 0, stream>>>(fW1, fW2, W1h, W2h);
  interp_kernel<<<dim3((NTOT + 255) / 256, 128), 256, 0, stream>>>(f1, idx3, w3, Xh);
  gemm_kernel<128, true ><<<dim3(NTOT / 32, 512 / 64), 256, 0, stream>>>(W1h, Xh, fb1, (void*)Y1h);
  gemm_kernel<512, false><<<dim3(NTOT / 32, 512 / 64), 256, 0, stream>>>(W2h, Y1h, fb2, d_out);
}
